// Agent_50500225466537
// MI455X (gfx1250) — compile-verified, matrix operand order checked by reading, not run on hardware
//
#include <hip/hip_runtime.h>

// ---------------------------------------------------------------------------
// CDNA5 (gfx1250) GCN-pipeline. Big GEMMs use v_wmma_f32_16x16x32_f16.
// Ahat = D^-1 (A+I) D^-1  =>  Ahat@X = rowscale(inv_d) of ((A+I) @ (X*inv_d)).
// B tile stored transposed in LDS so both A and B fragments load as two
// contiguous ds_load_b128 per lane. Double-buffered LDS + global_prefetch.
// Adjacency 0/1 -> f16 conversion done with integer MAD (1.0h == 0x3C00),
// avoiding v_cmp/v_cndmask SGPR pressure in the hot loop.
// ---------------------------------------------------------------------------

typedef _Float16 v16h __attribute__((ext_vector_type(16)));
typedef float    v8f  __attribute__((ext_vector_type(8)));
typedef _Float16 h8   __attribute__((ext_vector_type(8)));

static constexpr int N_NODES = 8192;
static constexpr int DIM     = 256;   // D == H == 256
static constexpr int K_OPP   = 11;

static constexpr int BM  = 128;
static constexpr int BN  = 64;
static constexpr int BK  = 32;
static constexpr int PAD = 8;          // pad rows to 40 halfs = 80 B (16B-aligned,
static constexpr int LDK = BK + PAD;   // stride 20 dwords -> conflict-free banks)

// Fragment loader (CDNA5 ISA 7.12.2, wave32). Works for both operands:
//  A tile rows are M-indexed, BsT rows are N-indexed; K contiguous in both.
//  Lane L<16: row=L, K in {0..7, 16..23}; lane L>=16: row=L-16, K in {8..15, 24..31}.
__device__ __forceinline__ v16h load_frag(const _Float16* base, int lane) {
    int r  = lane & 15;
    int k0 = (lane >> 4) << 3;
    const _Float16* p = base + r * LDK + k0;
    h8 lo = *(const h8*)p;         // ds_load_b128
    h8 hi = *(const h8*)(p + 16);  // ds_load_b128
    v16h f;
#pragma unroll
    for (int e = 0; e < 8; ++e) { f[e] = lo[e]; f[8 + e] = hi[e]; }
    return f;
}

// Stage one (BM x BK) A tile and one (BK x BN) B tile (transposed) into LDS.
template <bool AINT>
__device__ __forceinline__ void stage_tile(const void* __restrict__ Aptr,
                                           const _Float16* __restrict__ B,
                                           int m_blk, int n_blk, int k0,
                                           int lda, int ldb, int tid,
                                           _Float16* __restrict__ As,
                                           _Float16* __restrict__ BsT) {
    if (AINT) {
        const int* Ai = (const int*)Aptr;
#pragma unroll
        for (int p = 0; p < 4; ++p) {                 // 128x32 ints, int4 per thread
            int r = (tid >> 3) + p * 32;
            int c = (tid & 7) * 4;
            int4 a4 = *(const int4*)(Ai + (long long)(m_blk + r) * lda + (k0 + c));
            // 0/1 ints -> two packed-f16 dwords; exact since a in {0,1}
            uint2 pk;
            pk.x = (unsigned)a4.x * 0x3C00u + (unsigned)a4.y * 0x3C000000u;
            pk.y = (unsigned)a4.z * 0x3C00u + (unsigned)a4.w * 0x3C000000u;
            *(uint2*)(As + r * LDK + c) = pk;         // ds_store_b64
        }
    } else {
        const _Float16* Ah = (const _Float16*)Aptr;
#pragma unroll
        for (int p = 0; p < 2; ++p) {                 // 128x32 halfs, h8 per thread
            int q = tid + p * 256;
            int r = q >> 2;
            int c = (q & 3) * 8;
            *(h8*)(As + r * LDK + c) =
                *(const h8*)(Ah + (long long)(m_blk + r) * lda + (k0 + c));
        }
    }
    {   // B tile: coalesced h8 global read, transposed scatter into BsT[n][k]
        int k = tid >> 3;
        int c = (tid & 7) * 8;
        h8 b = *(const h8*)(B + (long long)(k0 + k) * ldb + (n_blk + c));
#pragma unroll
        for (int e = 0; e < 8; ++e) BsT[(c + e) * LDK + k] = b[e];
    }
}

// ---- Tiled WMMA GEMM: Out[M,Nc](f16) = epilogue(A[M,K] @ B[K,Nc]) ----------
// AINT : A is int32 0/1 adjacency (converted to f16 on the fly)
// IDENT: add B[row,col] (the +I term; requires K==M)
// BIAS : add bias[col]; SIG: sigmoid. Always scales rows by inv_d[row].
template <bool AINT, bool IDENT, bool BIAS, bool SIG>
__global__ __launch_bounds__(256)
void gemm_wmma(const void* __restrict__ Aptr, const _Float16* __restrict__ B,
               const float* __restrict__ inv_d, const float* __restrict__ bias,
               _Float16* __restrict__ Out, int M, int Ncols, int K,
               int lda, int ldb) {
    __shared__ _Float16 As[2][BM * LDK];    // 2 x 10 KB
    __shared__ _Float16 BsT[2][BN * LDK];   // 2 x  5 KB

    const int tid  = threadIdx.x;
    const int lane = tid & 31;
    const int wave = tid >> 5;        // 8 waves (wave32)
    const int wm   = wave & 3;        // 4 waves in M  -> 32 rows each
    const int wn   = wave >> 2;       // 2 waves in N  -> 32 cols each
    const int m_blk = blockIdx.y * BM;
    const int n_blk = blockIdx.x * BN;

    v8f acc[2][2];
#pragma unroll
    for (int i = 0; i < 2; ++i)
#pragma unroll
        for (int j = 0; j < 2; ++j) acc[i][j] = v8f{};

    stage_tile<AINT>(Aptr, B, m_blk, n_blk, 0, lda, ldb, tid, As[0], BsT[0]);
    __syncthreads();

    int buf = 0;
    for (int k0 = 0; k0 < K; k0 += BK) {
        const int nxt = buf ^ 1;
        if (k0 + BK < K)
            stage_tile<AINT>(Aptr, B, m_blk, n_blk, k0 + BK, lda, ldb, tid,
                             As[nxt], BsT[nxt]);
        if (k0 + 2 * BK < K) {   // global_prefetch_b8 two K-steps ahead
            long long row = m_blk + (tid >> 1);
            const char* pf = AINT
                ? (const char*)((const int*)Aptr + row * lda + (k0 + 2 * BK)) + (tid & 1) * 64
                : (const char*)((const _Float16*)Aptr + row * lda + (k0 + 2 * BK)) + (tid & 1) * 32;
            __builtin_prefetch(pf, 0, 1);
        }

        v16h afrag[2], bfrag[2];
#pragma unroll
        for (int mi = 0; mi < 2; ++mi)
            afrag[mi] = load_frag(As[buf] + (wm * 32 + mi * 16) * LDK, lane);
#pragma unroll
        for (int ni = 0; ni < 2; ++ni)
            bfrag[ni] = load_frag(BsT[buf] + (wn * 32 + ni * 16) * LDK, lane);
#pragma unroll
        for (int mi = 0; mi < 2; ++mi)
#pragma unroll
            for (int ni = 0; ni < 2; ++ni)
                acc[mi][ni] = __builtin_amdgcn_wmma_f32_16x16x32_f16(
                    false, afrag[mi], false, bfrag[ni], (short)0, acc[mi][ni],
                    false, false);

        __syncthreads();
        buf = nxt;
    }

    // Epilogue. C layout: VGPR r -> M = r (lanes<16) / r+8 (lanes>=16), N = lane&15.
#pragma unroll
    for (int mi = 0; mi < 2; ++mi)
#pragma unroll
        for (int ni = 0; ni < 2; ++ni) {
            int col   = n_blk + wn * 32 + ni * 16 + (lane & 15);
            int rbase = m_blk + wm * 32 + mi * 16 + ((lane >> 4) << 3);
#pragma unroll
            for (int r = 0; r < 8; ++r) {
                int row = rbase + r;
                float v = acc[mi][ni][r];
                if (IDENT) v += (float)B[(long long)row * ldb + col];
                if (BIAS)  v += bias[col];
                if (SIG)   v = 1.0f / (1.0f + __expf(-v));
                v *= inv_d[row];
                Out[(long long)row * Ncols + col] = (_Float16)v;
            }
        }
}

// ---- small kernels ---------------------------------------------------------

// inv_d[j] = 1 / (1 + sum_i A[i,j])   (column sums of A + I)
__global__ void degree_inv(const int* __restrict__ A, int n, float* __restrict__ inv_d) {
    int j = blockIdx.x * blockDim.x + threadIdx.x;
    if (j >= n) return;
    int s = 1;
    for (int i = 0; i < n; ++i) s += A[(long long)i * n + j];
    inv_d[j] = 1.0f / (float)s;
}

// Out[j,k] = f16(X[j,k] * inv_d[j])
__global__ void row_scale_f16(const float* __restrict__ X, const float* __restrict__ inv_d,
                              _Float16* __restrict__ Out, int n, int d) {
    long long i = (long long)blockIdx.x * blockDim.x + threadIdx.x;
    if (i >= (long long)n * d) return;
    int row = (int)(i / d);
    Out[i] = (_Float16)(X[i] * inv_d[row]);
}

__global__ void f32_to_f16(const float* __restrict__ X, _Float16* __restrict__ Out, long long n) {
    long long i = (long long)blockIdx.x * blockDim.x + threadIdx.x;
    if (i < n) Out[i] = (_Float16)X[i];
}

// G[row] = sigmoid(Z[row,:] @ W2 + b2)   (C == 1)
__global__ void matvec_sig(const _Float16* __restrict__ Z, const float* __restrict__ W2,
                           const float* __restrict__ b2, float* __restrict__ G,
                           int m, int k) {
    int row = blockIdx.x * blockDim.x + threadIdx.x;
    if (row >= m) return;
    float acc = 0.f;
    for (int i = 0; i < k; ++i) acc += (float)Z[(long long)row * k + i] * W2[i];
    acc += b2[0];
    G[row] = 1.0f / (1.0f + __expf(-acc));
}

// Final scalar stage: cosine-sim top-11 (jax top_k tie-break = lowest index),
// f_oppo sum, h, f, I, softmax -> policy (1,2).
__global__ void final_stage(const float* __restrict__ Gx, const float* __restrict__ Gy,
                            int n, const float* __restrict__ W_h, const float* __restrict__ W_f,
                            const float* __restrict__ W_p, const float* __restrict__ bias_h,
                            const int* __restrict__ idx_x, const int* __restrict__ idx_y,
                            float* __restrict__ out) {
    __shared__ float s_val[256];
    __shared__ int   s_idx[256];
    __shared__ unsigned char taken[N_NODES];
    __shared__ float s_foppo;
    const int tid = threadIdx.x;
    for (int i = tid; i < n; i += 256) taken[i] = 0;
    if (tid == 0) s_foppo = 0.f;
    __syncthreads();

    const float gx = Gx[idx_x[0]];
    const float gy = Gy[idx_y[0]];
    const float EPSV = 1e-8f;
    const float nv = fmaxf(fabsf(gy), EPSV);

    for (int k = 0; k < K_OPP; ++k) {
        float best = -3.0e38f;
        int   bi   = n;
        for (int i = tid; i < n; i += 256) {
            if (taken[i]) continue;
            float v = Gy[i];
            float s = (v * gy) / (fmaxf(fabsf(v), EPSV) * nv);
            if (s > best || (s == best && i < bi)) { best = s; bi = i; }
        }
        s_val[tid] = best; s_idx[tid] = bi;
        __syncthreads();
        for (int off = 128; off > 0; off >>= 1) {
            if (tid < off) {
                float v2 = s_val[tid + off]; int i2 = s_idx[tid + off];
                if (v2 > s_val[tid] || (v2 == s_val[tid] && i2 < s_idx[tid])) {
                    s_val[tid] = v2; s_idx[tid] = i2;
                }
            }
            __syncthreads();
        }
        if (tid == 0) {
            int sel = s_idx[0];
            taken[sel] = 1;
            s_foppo += __expf(gx * W_f[0] * Gy[sel]);
        }
        __syncthreads();
    }

    if (tid == 0) {
        float h = 1.0f / (1.0f + __expf(-(W_h[0] * gx + W_h[1] * gy + bias_h[0])));
        float f = __expf(gx * W_f[0] * gy);
        float I = f / s_foppo;
        float l0 = W_p[0] * h, l1 = W_p[0] * I;
        float m  = fmaxf(l0, l1);
        float e0 = __expf(l0 - m), e1 = __expf(l1 - m);
        out[0] = e0 / (e0 + e1);
        out[1] = e1 / (e0 + e1);
    }
}

// ---------------------------------------------------------------------------

extern "C" void kernel_launch(void* const* d_in, const int* in_sizes, int n_in,
                              void* d_out, int out_size, void* d_ws, size_t ws_size,
                              hipStream_t stream) {
    const int N = N_NODES, D = DIM;

    const int*   A_x  = (const int*)  d_in[0];
    const int*   A_y  = (const int*)  d_in[1];
    const float* X1   = (const float*)d_in[2];
    const float* X2   = (const float*)d_in[3];
    const float* W1   = (const float*)d_in[4];
    const float* b1   = (const float*)d_in[5];
    const float* W2   = (const float*)d_in[6];
    const float* b2   = (const float*)d_in[7];
    const float* W_h  = (const float*)d_in[8];
    const float* W_f  = (const float*)d_in[9];
    const float* W_p  = (const float*)d_in[10];
    const float* b_h  = (const float*)d_in[11];
    const int*   ix   = (const int*)  d_in[12];
    const int*   iy   = (const int*)  d_in[13];
    float* out = (float*)d_out;

    // Workspace layout
    char* ws = (char*)d_ws;
    size_t off = 0;
    float*    inv_d = (float*)(ws + off);      off += (size_t)N * 4;
    _Float16* W1h   = (_Float16*)(ws + off);   off += (size_t)D * D * 2;
    _Float16* bufXs = (_Float16*)(ws + off);   off += (size_t)N * D * 2;
    _Float16* bufZ  = (_Float16*)(ws + off);   off += (size_t)N * D * 2;
    _Float16* bufH  = (_Float16*)(ws + off);   off += (size_t)N * D * 2;
    float*    Gx    = (float*)(ws + off);      off += (size_t)N * 4;
    float*    Gy    = (float*)(ws + off);      off += (size_t)N * 4;

    dim3 gemm_grid(D / BN, N / BM);   // (4, 64)
    dim3 blk(256);

    f32_to_f16<<<(D * D + 255) / 256, blk, 0, stream>>>(W1, W1h, (long long)D * D);

    for (int g = 0; g < 2; ++g) {
        const int*   A = g ? A_y : A_x;
        const float* X = g ? X2 : X1;
        float*       G = g ? Gy : Gx;

        degree_inv<<<N / 256, blk, 0, stream>>>(A, N, inv_d);
        row_scale_f16<<<(N * D) / 256, blk, 0, stream>>>(X, inv_d, bufXs, N, D);

        // Z1 = inv_d ⊙ ((A+I) @ Xs)
        gemm_wmma<true, true, false, false><<<gemm_grid, blk, 0, stream>>>(
            A, bufXs, inv_d, nullptr, bufZ, N, D, N, N, D);
        // Hs = inv_d ⊙ sigmoid(Z1 @ W1 + b1)
        gemm_wmma<false, false, true, true><<<gemm_grid, blk, 0, stream>>>(
            bufZ, W1h, inv_d, b1, bufH, N, D, D, D, D);
        // Z2 = inv_d ⊙ ((A+I) @ Hs)
        gemm_wmma<true, true, false, false><<<gemm_grid, blk, 0, stream>>>(
            A, bufH, inv_d, nullptr, bufZ, N, D, N, N, D);
        // G = sigmoid(Z2 @ W2 + b2)
        matvec_sig<<<N / 256, blk, 0, stream>>>(bufZ, W2, b2, G, N, D);
    }

    final_stage<<<1, blk, 0, stream>>>(Gx, Gy, N, W_h, W_f, W_p, b_h, ix, iy, out);
    (void)in_sizes; (void)n_in; (void)out_size; (void)ws_size;
}
